// GINEConvGraphGymLayer_24902220383106
// MI455X (gfx1250) — matrix operand order repaired and checked by reading, not compiled, not run on hardware
//
#include <hip/hip_runtime.h>

typedef __bf16 bf16_t;
typedef __attribute__((ext_vector_type(16))) __bf16 v16bf;
typedef __attribute__((ext_vector_type(8)))  __bf16 v8bf;
typedef __attribute__((ext_vector_type(8)))  float   v8f;
typedef __attribute__((ext_vector_type(4)))  unsigned int u32x4;
typedef __attribute__((ext_vector_type(8)))  int         i32x8;
typedef __attribute__((ext_vector_type(4)))  int         i32x4;

constexpr int kD = 128;        // feature dim
constexpr int kBM = 80;        // rows per block (100000 = 80 * 1250)
constexpr float kEps = 1e-5f;

#if defined(__HIP_DEVICE_COMPILE__) && defined(__gfx1250__) && \
    __has_builtin(__builtin_amdgcn_tensor_load_to_lds)
#define USE_TDM 1
#else
#define USE_TDM 0
#endif

// ---------------------------------------------------------------------------
// prep: zero aggregation buffer (aliased onto d_out), zero BN sums, and write
// W1^T as bf16 (WbT[n*128+k]) so the GEMM weight stage is a contiguous copy.
// ---------------------------------------------------------------------------
__global__ void prep_kernel(const float* __restrict__ W1,
                            float* __restrict__ aggr,
                            float* __restrict__ sums,
                            bf16_t* __restrict__ W1bT, int nd) {
  int g = blockIdx.x * blockDim.x + threadIdx.x;
  int stride = gridDim.x * blockDim.x;
  for (int i = g; i < nd; i += stride) aggr[i] = 0.0f;
  for (int i = g; i < kD * kD; i += stride) {
    int k = i >> 7, n = i & 127;
    W1bT[n * kD + k] = (bf16_t)W1[i];
  }
  for (int i = g; i < 2 * kD; i += stride) sums[i] = 0.0f;
}

// ---------------------------------------------------------------------------
// scatter: msg = relu(x[src] + edge_attr); global_atomic_add_f32 into
// aggr[dst] (aggr is 51 MB -> atomics resolve in the 192 MB L2).
// One thread = 4 contiguous features of one edge (float4 loads, coalesced).
// ---------------------------------------------------------------------------
__global__ __launch_bounds__(256) void scatter_kernel(
    const float* __restrict__ x, const int* __restrict__ ei,
    const float* __restrict__ ea, float* __restrict__ aggr, int E) {
  long long g = (long long)blockIdx.x * 256 + threadIdx.x;
  int e = (int)(g >> 5);
  if (e >= E) return;
  int c = ((int)(g & 31)) << 2;
  int src = ei[e];
  int dst = ei[E + e];
  const float4 xv = *(const float4*)(x + (size_t)src * kD + c);
  const float4 av = *(const float4*)(ea + (size_t)e * kD + c);
  float m0 = fmaxf(xv.x + av.x, 0.0f);
  float m1 = fmaxf(xv.y + av.y, 0.0f);
  float m2 = fmaxf(xv.z + av.z, 0.0f);
  float m3 = fmaxf(xv.w + av.w, 0.0f);
  float* p = aggr + (size_t)dst * kD + c;
  atomicAdd(p + 0, m0);
  atomicAdd(p + 1, m1);
  atomicAdd(p + 2, m2);
  atomicAdd(p + 3, m3);
}

// ---------------------------------------------------------------------------
// GEMM: [80 rows] x [128x128], 8 wave32 per block. Each wave owns a 16-wide
// column strip and 5 16x16 M-tiles -> 20x v_wmma_f32_16x16x32_bf16 per wave
// per staged weight. Weight (32 KB, pre-transposed bf16) is staged into LDS
// by the Tensor Data Mover (tensor_load_to_lds + s_wait_tensorcnt) while the
// 256 threads convert/stage the A tile.
// MODE 0: A = bf16(x + aggr); epilogue relu, h1(bf16) store, BN sum/sumsq.
// MODE 1: A = h1 (bf16); BN folded into weights/bias; f32 output store.
// ---------------------------------------------------------------------------
template <int MODE>
__global__ __launch_bounds__(256) void gemm_kernel(
    const float* __restrict__ xin, const float* __restrict__ aggr,
    const bf16_t* __restrict__ h1in, const bf16_t* __restrict__ WbT,
    const float* __restrict__ bias, bf16_t* __restrict__ h1out,
    float* __restrict__ sums, float* __restrict__ out, int Nrows) {
  __shared__ bf16_t sW[kD * kD];    // W^T: sW[n*128 + k]
  __shared__ bf16_t sA[kBM * kD];   // A tile row-major

  const int tid = threadIdx.x;
  const int m_base = blockIdx.x * kBM;

  // ---- stage W^T (contiguous 32 KB) via TDM; fallback: vector copy --------
#if USE_TDM
  if (tid < 32) {  // single wave issues the DMA (EXEC-independent)
    unsigned long long ga = (unsigned long long)(uintptr_t)WbT;
    unsigned int la = (unsigned int)(uintptr_t)&sW[0];
    u32x4 g0;
    g0[0] = 1u;                                   // count=1 (valid user D#)
    g0[1] = la;                                   // lds_addr (bytes)
    g0[2] = (unsigned int)ga;                     // global_addr[31:0]
    g0[3] = (unsigned int)((ga >> 32) & 0x1FFFFFFu) | (2u << 30);  // [56:32], type=2
    i32x8 g1;
    g1[0] = (int)(3u << 16);                      // data_size = 8 bytes
    g1[1] = (int)((4096u & 0xFFFFu) << 16);       // tensor_dim0[15:0] = 4096
    g1[2] = (int)(4096u >> 16);                   // tensor_dim0[31:16], dim1=0
    g1[3] = (int)((4096u & 0xFFFFu) << 16);       // tile_dim0 = 4096 (1-D tile)
    g1[4] = 0;                                    // tile_dim1/2 unused
    g1[5] = 0; g1[6] = 0; g1[7] = 0;              // strides unused (1-D)
    i32x4 g2 = {0, 0, 0, 0};
    i32x4 g3 = {0, 0, 0, 0};
    i32x8 g4 = {0, 0, 0, 0, 0, 0, 0, 0};
    __builtin_amdgcn_tensor_load_to_lds(g0, g1, g2, g3, g4, 0);
    __builtin_amdgcn_s_wait_tensorcnt(0);
  }
#else
  for (int i = tid; i < kD * kD / 8; i += 256)
    ((v8bf*)sW)[i] = ((const v8bf*)WbT)[i];
#endif

  // ---- stage A tile (80 x 128), vectorized 8-wide ------------------------
  for (int i = tid; i < kBM * kD / 8; i += 256) {
    int base = i * 8;
    int r = base >> 7;
    int col = base & 127;
    int row = m_base + r;
    v8bf a8;
    if (MODE == 0) {
      if (row < Nrows) {
        const float4 x0 = *(const float4*)(xin + (size_t)row * kD + col);
        const float4 x1 = *(const float4*)(xin + (size_t)row * kD + col + 4);
        const float4 a0 = *(const float4*)(aggr + (size_t)row * kD + col);
        const float4 a1 = *(const float4*)(aggr + (size_t)row * kD + col + 4);
        a8[0] = (bf16_t)(x0.x + a0.x);
        a8[1] = (bf16_t)(x0.y + a0.y);
        a8[2] = (bf16_t)(x0.z + a0.z);
        a8[3] = (bf16_t)(x0.w + a0.w);
        a8[4] = (bf16_t)(x1.x + a1.x);
        a8[5] = (bf16_t)(x1.y + a1.y);
        a8[6] = (bf16_t)(x1.z + a1.z);
        a8[7] = (bf16_t)(x1.w + a1.w);
      } else {
#pragma unroll
        for (int j = 0; j < 8; ++j) a8[j] = (bf16_t)0.0f;
      }
    } else {
      if (row < Nrows) {
        a8 = *(const v8bf*)(h1in + (size_t)row * kD + col);
      } else {
#pragma unroll
        for (int j = 0; j < 8; ++j) a8[j] = (bf16_t)0.0f;
      }
    }
    *(v8bf*)&sA[base] = a8;
  }
  __syncthreads();

  const int wave = tid >> 5;
  const int lane = tid & 31;
  const int r = lane & 15;     // A row within sub-tile / output column in strip
  const int half = lane >> 4;  // lane-half per ISA WMMA VGPR layouts
  const int n0 = wave << 4;    // this wave's output-column base

  const float bv = bias[n0 + r];
  v8f acc[5];
#pragma unroll
  for (int t = 0; t < 5; ++t)
#pragma unroll
    for (int v = 0; v < 8; ++v) acc[t][v] = bv;

#pragma unroll
  for (int k0 = 0; k0 < kD; k0 += 32) {
    // B fragment: lane column n0+r, K = k0 + half*16 + [0..15], contiguous
    // 32 B in LDS because W is stored transposed.
    v16bf b = *(const v16bf*)&sW[(n0 + r) * kD + k0 + half * 16];
#pragma unroll
    for (int t = 0; t < 5; ++t) {
      // A fragment: VGPR0-3 hold K = k0 + half*16 + [0..7],
      //             VGPR4-7 hold K = k0 + half*16 + 8 + [0..7]
      v8bf al = *(const v8bf*)&sA[(t * 16 + r) * kD + k0 + half * 16];
      v8bf ah = *(const v8bf*)&sA[(t * 16 + r) * kD + k0 + half * 16 + 8];
      v16bf a;
#pragma unroll
      for (int j = 0; j < 8; ++j) {
        a[j] = al[j];
        a[j + 8] = ah[j];
      }
      acc[t] = __builtin_amdgcn_wmma_f32_16x16x32_bf16(
          false, a, false, b, (short)0, acc[t], false, false);
    }
  }

  // C/D layout: lane column = n0 + r; rows = m_base + t*16 + half*8 + v
  if (MODE == 0) {
    float s = 0.0f, s2 = 0.0f;
#pragma unroll
    for (int t = 0; t < 5; ++t) {
#pragma unroll
      for (int v = 0; v < 8; ++v) {
        int m = m_base + t * 16 + half * 8 + v;
        if (m < Nrows) {
          float f = fmaxf(acc[t][v], 0.0f);
          h1out[(size_t)m * kD + n0 + r] = (bf16_t)f;
          s += f;
          s2 += f * f;
        }
      }
    }
    atomicAdd(&sums[n0 + r], s);
    atomicAdd(&sums[kD + n0 + r], s2);
  } else {
#pragma unroll
    for (int t = 0; t < 5; ++t) {
#pragma unroll
      for (int v = 0; v < 8; ++v) {
        int m = m_base + t * 16 + half * 8 + v;
        if (m < Nrows) out[(size_t)m * kD + n0 + r] = acc[t][v];
      }
    }
  }
}

// ---------------------------------------------------------------------------
// fold BN into the second linear layer (writes W2'^T bf16 + folded bias):
//   s[k] = gamma[k]*rsqrt(var[k]+eps);  t[k] = beta[k] - mean[k]*s[k]
//   W2T'[n][k] = bf16(s[k]*W2[k][n]);   b2'[n] = b2[n] + sum_k t[k]*W2[k][n]
// ---------------------------------------------------------------------------
__global__ __launch_bounds__(128) void fold_kernel(
    const float* __restrict__ sums, const float* __restrict__ gamma,
    const float* __restrict__ beta, const float* __restrict__ W2,
    const float* __restrict__ b2, bf16_t* __restrict__ W2bT,
    float* __restrict__ b2p, float invN) {
  __shared__ float s_sh[kD];
  __shared__ float t_sh[kD];
  int n = threadIdx.x;
  float mean = sums[n] * invN;
  float var = sums[kD + n] * invN - mean * mean;
  float s = gamma[n] * rsqrtf(var + kEps);
  float t = beta[n] - mean * s;
  s_sh[n] = s;
  t_sh[n] = t;
  __syncthreads();
  float bb = b2[n];
  for (int k = 0; k < kD; ++k) {
    float w = W2[k * kD + n];
    W2bT[n * kD + k] = (bf16_t)(s_sh[k] * w);  // contiguous per-thread writes
    bb += t_sh[k] * w;
  }
  b2p[n] = bb;
}

// ---------------------------------------------------------------------------
extern "C" void kernel_launch(void* const* d_in, const int* in_sizes, int n_in,
                              void* d_out, int out_size, void* d_ws,
                              size_t ws_size, hipStream_t stream) {
  const float* x = (const float*)d_in[0];
  const int* ei = (const int*)d_in[1];
  const float* ea = (const float*)d_in[2];
  const float* W1 = (const float*)d_in[3];
  const float* b1 = (const float*)d_in[4];
  const float* gamma = (const float*)d_in[5];
  const float* beta = (const float*)d_in[6];
  const float* W2 = (const float*)d_in[7];
  const float* b2 = (const float*)d_in[8];

  const int N = in_sizes[0] / kD;
  const int E = in_sizes[2] / kD;
  float* out = (float*)d_out;

  // workspace layout
  char* ws = (char*)d_ws;
  bf16_t* h1 = (bf16_t*)ws;
  size_t off = (size_t)N * kD * sizeof(bf16_t);
  bf16_t* W1bT = (bf16_t*)(ws + off); off += (size_t)kD * kD * sizeof(bf16_t);
  bf16_t* W2bT = (bf16_t*)(ws + off); off += (size_t)kD * kD * sizeof(bf16_t);
  float* sums = (float*)(ws + off);   off += 2 * kD * sizeof(float);
  float* b2p = (float*)(ws + off);

  float* aggr = out;  // reuse d_out as scatter target; overwritten by gemm2

  prep_kernel<<<1024, 256, 0, stream>>>(W1, aggr, sums, W1bT, N * kD);

  long long tot = (long long)E * 32;
  int sblocks = (int)((tot + 255) / 256);
  scatter_kernel<<<sblocks, 256, 0, stream>>>(x, ei, ea, aggr, E);

  int gblocks = (N + kBM - 1) / kBM;
  gemm_kernel<0><<<gblocks, 256, 0, stream>>>(x, aggr, nullptr, W1bT, b1, h1,
                                              sums, nullptr, N);

  fold_kernel<<<1, 128, 0, stream>>>(sums, gamma, beta, W2, b2, W2bT, b2p,
                                     1.0f / (float)N);

  gemm_kernel<1><<<gblocks, 256, 0, stream>>>(nullptr, nullptr, h1, W2bT, b2p,
                                              nullptr, nullptr, out, N);
}
